// MyModel_87522843560931
// MI455X (gfx1250) — matrix operand order repaired and checked by reading, not compiled, not run on hardware
//
#include <hip/hip_runtime.h>

// ---------------------------------------------------------------------------
// SimpleRNN scan fused kernel for MI455X (gfx1250, wave32, WMMA).
//
//   h_t = tanh([x_t | h_{t-1}] @ [Wk ; Wr])   (K = 256+512 = 768, N = 512)
//   out = h_T @ Wd
//
// Grid: 16 workgroups (one per 16-row batch block), 256 threads (8 wave32).
// Each wave computes a 16x64 slice of h_t via v_wmma_f32_16x16x32_bf16.
// Weights pre-converted to bf16, transposed ([N][K] row-major) in d_ws so
// B-matrix fragments are contiguous 32B loads (L2-resident, ~1 MB total).
// No cross-workgroup synchronization: batch rows are independent through the
// whole recurrence.
//
// NOTE: a zero SGPR offset laundered through an empty asm is added to the
// weight address each timestep. This (a) blocks LICM from hoisting the
// t-invariant weight loads out of the serial loop (which spills ~3KB/lane
// to scratch), while (b) preserving global address-space provenance so the
// loads stay global_load_b128 (flat_load couples LOADcnt with DScnt).
// ---------------------------------------------------------------------------

typedef __bf16 bf16_t;
typedef __attribute__((ext_vector_type(16))) __bf16 v16bf;
typedef __attribute__((ext_vector_type(8)))  __bf16 v8bf;
typedef __attribute__((ext_vector_type(8)))  float  v8f;

#define B_    256
#define T_    1024
#define D_    256
#define U_    512
#define O_    256
#define BBLK  16
#define KTOT  768            // D + U (concatenated GEMM K)
#define XS_STRIDE 264        // 256 + 8 pad  (528B rows, 16B aligned, bank-spread)
#define HS_STRIDE 520        // 512 + 8 pad  (1040B rows, 16B aligned, bank-spread)

#define WT_ELEMS  (512 * 768)    // [Wk;Wr]^T bf16 in ws
#define WDT_ELEMS (256 * 512)    // Wd^T bf16 in ws

#define WMMA_BF16 __builtin_amdgcn_wmma_f32_16x16x32_bf16

__device__ __forceinline__ float tanh_fast(float x) {
  // tanh(x) = sign(x) * (1 - e) / (1 + e),  e = exp(-2|x|)  via v_exp_f32
  float ax = __builtin_fabsf(x);
  float e  = __builtin_amdgcn_exp2f(ax * -2.8853900817779268f); // -2/ln2
  float r  = (1.0f - e) * __builtin_amdgcn_rcpf(1.0f + e);
  return x < 0.0f ? -r : r;
}

__device__ __forceinline__ v16bf cat8(v8bf lo, v8bf hi) {
  return __builtin_shufflevector(lo, hi, 0,1,2,3,4,5,6,7,8,9,10,11,12,13,14,15);
}

// A-fragment load (contiguous k order, matched by the B loads): lane holds row (lane&15); two
// contiguous 16B chunks at K = k0+16*half and K = k0+16*half+8.
__device__ __forceinline__ v16bf load_afrag(const bf16_t* row_khalf, int k0) {
  v8bf lo = *(const v8bf*)(row_khalf + k0);
  v8bf hi = *(const v8bf*)(row_khalf + k0 + 8);
  return cat8(lo, hi);
}

// Convert fp32 weights -> bf16, transposed, into workspace.
// ws[0 .. WT_ELEMS)          : WT [n=512][k=768], WT[n][k] = (k<256? Wk[k][n] : Wr[k-256][n])
// ws[WT_ELEMS .. +WDT_ELEMS) : WDT [n=256][k=512], WDT[n][k] = Wd[k][n]
__global__ void convert_weights_kernel(const float* __restrict__ Wk,
                                       const float* __restrict__ Wr,
                                       const float* __restrict__ Wd,
                                       bf16_t* __restrict__ ws) {
  int idx = blockIdx.x * blockDim.x + threadIdx.x;
  if (idx < WT_ELEMS) {
    int n = idx / KTOT, k = idx % KTOT;
    float v = (k < D_) ? Wk[(size_t)k * U_ + n]
                       : Wr[(size_t)(k - D_) * U_ + n];
    ws[idx] = (bf16_t)v;
  } else {
    int j = idx - WT_ELEMS;
    if (j < WDT_ELEMS) {
      int n = j / U_, k = j % U_;
      ws[WT_ELEMS + j] = (bf16_t)Wd[(size_t)k * O_ + n];
    }
  }
}

__global__ __launch_bounds__(256, 1)
void rnn_scan_kernel(const float*  __restrict__ x,    // [B,T,D] fp32
                     const float*  __restrict__ h0,   // [B,U]   fp32
                     const bf16_t* __restrict__ wt,   // [512][768] bf16
                     const bf16_t* __restrict__ wdt,  // [256][512] bf16
                     float*        __restrict__ out)  // [B,O]   fp32
{
  __shared__ bf16_t xs[BBLK][XS_STRIDE];        // x_t tile, bf16
  __shared__ bf16_t hs[2][BBLK][HS_STRIDE];     // double-buffered h tile, bf16

  const int tid  = threadIdx.x;
  const int lane = tid & 31;
  const int wave = tid >> 5;        // 0..7
  const int lan  = lane & 15;       // M index for A-frag / N index for B-frag
  const int half = lane >> 4;       // K-half selector (WMMA 16-bit layout)
  const int b0   = blockIdx.x * BBLK;

  // --- init h[0] from h0 (fp32 -> bf16): 16x512, 32 elems per thread ------
  {
    int r  = tid >> 4;              // 0..15
    int c0 = (tid & 15) * 32;       // 0..480
    const float* hp = h0 + (size_t)(b0 + r) * U_ + c0;
#pragma unroll
    for (int c = 0; c < 32; ++c)
      hs[0][r][c0 + c] = (bf16_t)hp[c];
  }

  const int nbase = wave * 64;      // this wave's 64 output columns of h

  // Hoisted per-lane base pointers.
  const bf16_t* xrow  = &xs[lan][16 * half];               // x part of A
  const bf16_t* hrowA = &hs[0][lan][16 * half];            // h part of A, buf 0
  const bf16_t* hrowB = &hs[1][lan][16 * half];            // h part of A, buf 1
  const bf16_t* wbase = wt + (size_t)(nbase + lan) * KTOT + 16 * half;

  int cur = 0;

  for (int t = 0; t < T_; ++t) {
    __syncthreads();                // h[cur] ready; safe to overwrite xs

    // --- stage x_t block [16 x 256] fp32 -> bf16 LDS --------------------
    {
      int r  = tid >> 4;
      int c0 = (tid & 15) * 16;
      const float* xp = x + ((size_t)(b0 + r) * T_ + t) * D_ + c0;
      float4 f0 = ((const float4*)xp)[0];
      float4 f1 = ((const float4*)xp)[1];
      float4 f2 = ((const float4*)xp)[2];
      float4 f3 = ((const float4*)xp)[3];
      bf16_t* dst = &xs[r][c0];
      dst[ 0]=(bf16_t)f0.x; dst[ 1]=(bf16_t)f0.y; dst[ 2]=(bf16_t)f0.z; dst[ 3]=(bf16_t)f0.w;
      dst[ 4]=(bf16_t)f1.x; dst[ 5]=(bf16_t)f1.y; dst[ 6]=(bf16_t)f1.z; dst[ 7]=(bf16_t)f1.w;
      dst[ 8]=(bf16_t)f2.x; dst[ 9]=(bf16_t)f2.y; dst[10]=(bf16_t)f2.z; dst[11]=(bf16_t)f2.w;
      dst[12]=(bf16_t)f3.x; dst[13]=(bf16_t)f3.y; dst[14]=(bf16_t)f3.z; dst[15]=(bf16_t)f3.w;
      if (t + 1 < T_) __builtin_prefetch(xp + D_, 0, 0);   // global_prefetch next x_t
    }
    __syncthreads();

    // Opaque zero offset (SGPR): blocks cross-t hoisting of weight loads
    // while keeping the address chain rooted at the global pointer `wt`.
    int woff = 0;
    asm volatile("" : "+s"(woff));
    const bf16_t* wrow = wbase + woff;

    v8f acc0 = {}, acc1 = {}, acc2 = {}, acc3 = {};

    // --- x_t (16 x 256) @ Wk^T slice: kc 0..7, A from xs ----------------
    for (int kc = 0; kc < D_ / 32; ++kc) {
      int k0 = kc * 32;
      v16bf a = load_afrag(xrow, k0);
      const bf16_t* bb = wrow + k0;
      v16bf bv0 = *(const v16bf*)(bb + (size_t)( 0) * KTOT);
      v16bf bv1 = *(const v16bf*)(bb + (size_t)(16) * KTOT);
      v16bf bv2 = *(const v16bf*)(bb + (size_t)(32) * KTOT);
      v16bf bv3 = *(const v16bf*)(bb + (size_t)(48) * KTOT);
      acc0 = WMMA_BF16(false, a, false, bv0, (short)0, acc0, false, false);
      acc1 = WMMA_BF16(false, a, false, bv1, (short)0, acc1, false, false);
      acc2 = WMMA_BF16(false, a, false, bv2, (short)0, acc2, false, false);
      acc3 = WMMA_BF16(false, a, false, bv3, (short)0, acc3, false, false);
    }

    // --- h_{t-1} (16 x 512) @ Wr^T slice: kc 0..15, A from hs[cur] ------
    const bf16_t* hrow = cur ? hrowB : hrowA;
    for (int kc = 0; kc < U_ / 32; ++kc) {
      int k0 = kc * 32;
      v16bf a = load_afrag(hrow, k0);
      const bf16_t* bb = wrow + D_ + k0;
      v16bf bv0 = *(const v16bf*)(bb + (size_t)( 0) * KTOT);
      v16bf bv1 = *(const v16bf*)(bb + (size_t)(16) * KTOT);
      v16bf bv2 = *(const v16bf*)(bb + (size_t)(32) * KTOT);
      v16bf bv3 = *(const v16bf*)(bb + (size_t)(48) * KTOT);
      acc0 = WMMA_BF16(false, a, false, bv0, (short)0, acc0, false, false);
      acc1 = WMMA_BF16(false, a, false, bv1, (short)0, acc1, false, false);
      acc2 = WMMA_BF16(false, a, false, bv2, (short)0, acc2, false, false);
      acc3 = WMMA_BF16(false, a, false, bv3, (short)0, acc3, false, false);
    }

    // --- tanh + write new h tile (C layout: VGPR r -> row r + 8*half) ---
    int nxt = cur ^ 1;
#pragma unroll
    for (int r = 0; r < 8; ++r) {
      int row = r + 8 * half;
      hs[nxt][row][nbase +  0 + lan] = (bf16_t)tanh_fast(acc0[r]);
      hs[nxt][row][nbase + 16 + lan] = (bf16_t)tanh_fast(acc1[r]);
      hs[nxt][row][nbase + 32 + lan] = (bf16_t)tanh_fast(acc2[r]);
      hs[nxt][row][nbase + 48 + lan] = (bf16_t)tanh_fast(acc3[r]);
    }
    cur = nxt;
  }

  __syncthreads();

  // --- head: out[16 x 256] = h_T @ Wd ; wave w -> cols [32w, 32w+32) -----
  {
    const int pb = wave * 32;
    const bf16_t* hrow = cur ? hrowB : hrowA;
    const bf16_t* wdrow = wdt + (size_t)(pb + lan) * U_ + 16 * half;
    v8f p0 = {}, p1 = {};
    for (int kc = 0; kc < U_ / 32; ++kc) {
      int k0 = kc * 32;
      v16bf a = load_afrag(hrow, k0);
      v16bf bv0 = *(const v16bf*)(wdrow + k0);
      v16bf bv1 = *(const v16bf*)(wdrow + k0 + (size_t)16 * U_);
      p0 = WMMA_BF16(false, a, false, bv0, (short)0, p0, false, false);
      p1 = WMMA_BF16(false, a, false, bv1, (short)0, p1, false, false);
    }
#pragma unroll
    for (int r = 0; r < 8; ++r) {
      int row = b0 + r + 8 * half;
      out[(size_t)row * O_ + pb +  0 + lan] = p0[r];
      out[(size_t)row * O_ + pb + 16 + lan] = p1[r];
    }
  }
}

extern "C" void kernel_launch(void* const* d_in, const int* in_sizes, int n_in,
                              void* d_out, int out_size, void* d_ws, size_t ws_size,
                              hipStream_t stream) {
  (void)in_sizes; (void)n_in; (void)out_size; (void)ws_size;
  const float* x  = (const float*)d_in[0];
  const float* Wk = (const float*)d_in[1];
  const float* Wr = (const float*)d_in[2];
  const float* Wd = (const float*)d_in[3];
  const float* h0 = (const float*)d_in[4];

  bf16_t* wsb = (bf16_t*)d_ws;   // ~1 MB bf16 weights (L2-resident)

  int total = WT_ELEMS + WDT_ELEMS;
  convert_weights_kernel<<<(total + 255) / 256, 256, 0, stream>>>(Wk, Wr, Wd, wsb);

  rnn_scan_kernel<<<B_ / BBLK, 256, 0, stream>>>(
      x, h0, wsb, wsb + WT_ELEMS, (float*)d_out);
}
